// GraphCastNet_26044681683202
// MI455X (gfx1250) — matrix operand order repaired and checked by reading, not compiled, not run on hardware
//
#include <hip/hip_runtime.h>
#include <hip/hip_bf16.h>

// ---------------------------------------------------------------------------
// CDNA5 (gfx1250) fused GraphCast MLP kernels using v_wmma_f32_16x16x32_bf16.
// BM=64 rows/block: each wave computes 4 M-tiles x 4 N-tiles; B operand
// register-cached and reused across the 4 M-tiles (4x weight-traffic cut).
// LayerNorm fused via register-level half-wave shuffle reductions.
// ---------------------------------------------------------------------------

typedef __attribute__((ext_vector_type(16))) __bf16 v16bf;
typedef __attribute__((ext_vector_type(8)))  __bf16 v8bf;
typedef __attribute__((ext_vector_type(8)))  float  v8f;

union ABvec { v16bf v; v8bf h[2]; };

__device__ __forceinline__ unsigned short f2bf(float f) {
    union { float f; unsigned u; } c; c.f = f;
    unsigned r = c.u + 0x7FFFu + ((c.u >> 16) & 1u);   // round-to-nearest-even
    return (unsigned short)(r >> 16);
}

__device__ __forceinline__ v8f zero8() {
    v8f z;
#pragma unroll
    for (int i = 0; i < 8; ++i) z[i] = 0.0f;
    return z;
}

// ---------------------------------------------------------------------------
// Weight repack: fp32 [K,N] row-major -> bf16 tiled [Kp/32][Np/16][32 lanes][16]
// as WMMA B (32x16) per lane in contiguous k order, k = 16*(lane/16) + j, same as the A loads.
// ---------------------------------------------------------------------------
__global__ void pack_w(const float* __restrict__ W, unsigned short* __restrict__ out,
                       int K, int N, int Kp, int Np) {
    size_t total = (size_t)Kp * (size_t)Np;
    int nt_cnt = Np >> 4;
    for (size_t i = (size_t)blockIdx.x * blockDim.x + threadIdx.x; i < total;
         i += (size_t)gridDim.x * blockDim.x) {
        int j    = (int)(i & 15);
        int lane = (int)((i >> 4) & 31);
        size_t t = i >> 9;
        int nt = (int)(t % nt_cnt);
        int kt = (int)(t / nt_cnt);
        int k = kt * 32 + (lane >> 4) * 16 + j;
        int n = nt * 16 + (lane & 15);
        float v = (k < K && n < N) ? W[(size_t)k * N + n] : 0.0f;
        out[i] = f2bf(v);
    }
}

__global__ void fill0(float* __restrict__ p, size_t n) {
    for (size_t i = (size_t)blockIdx.x * blockDim.x + threadIdx.x; i < n;
         i += (size_t)gridDim.x * blockDim.x) p[i] = 0.0f;
}

// ---------------------------------------------------------------------------
struct MlpArgs {
    const float* s0; const int* i0; int w0;
    const float* s1; const int* i1; int w1;
    const float* s2; const int* i2; int w2;
    int M;            // rows
    int Ktot;         // w0+w1+w2 (unpadded)
    int Kp;           // K padded to multiple of 32
    const unsigned short* w1p; const float* b1;
    const unsigned short* w2p; const float* b2;
    const float* gamma; const float* beta;      // null => no LayerNorm
    const float* res;                           // null => no residual
    float* outp; int Nout; int outStride;       // outp null => skip store
    float* aggp; const int* aggIdx;             // aggp null => no aggregation
    int N2p;          // GEMM2 padded N (512 or 240)
    int N2real;       // GEMM2 real  N (512 or 227)
};

#define BM  64
#define LDA 136   // lds_a K stride (ushorts), 16B aligned
#define LDH 520   // lds_h stride

__global__ __launch_bounds__(256) void mlp_fused(MlpArgs A) {
    __shared__ unsigned short lds_a[BM * LDA];   // 17.4 KB
    __shared__ unsigned short lds_h[BM * LDH];   // 66.6 KB
    __shared__ float          red_s[BM * 8];     //  2 KB
    __shared__ float          red_ss[BM * 8];    //  2 KB
    __shared__ float          st_mean[BM];
    __shared__ float          st_rstd[BM];

    const int tid  = threadIdx.x;
    const int lane = tid & 31;
    const int wave = tid >> 5;
    const int hi   = lane >> 4;
    const int lr   = lane & 15;
    const int rowBase = blockIdx.x * BM;
    const int off01 = A.w0 + A.w1;

    // ---------------- GEMM1: H = silu(concat(A) @ W1 + b1) ----------------
    v8f acc1[4][4];  // [M-tile][N-tile]
#pragma unroll
    for (int mt = 0; mt < 4; ++mt)
#pragma unroll
        for (int t = 0; t < 4; ++t) acc1[mt][t] = zero8();

    const int NT1 = 32;  // N = 512
    for (int k0 = 0; k0 < A.Kp; k0 += 128) {
        __syncthreads();
        // cooperative concat-gather load of A chunk (64 rows x 128 K), fp32->bf16
        for (int i = tid; i < BM * 128; i += 256) {
            int r = i >> 7, c = i & 127;
            int k = k0 + c;
            float v = 0.0f;
            int mm = rowBase + r;
            if (mm < A.M && k < A.Ktot) {
                const float* bp; const int* ip; int col, w;
                if (k < A.w0)       { bp = A.s0; ip = A.i0; col = k;          w = A.w0; }
                else if (k < off01) { bp = A.s1; ip = A.i1; col = k - A.w0;   w = A.w1; }
                else                { bp = A.s2; ip = A.i2; col = k - off01;  w = A.w2; }
                int rr = ip ? ip[mm] : mm;
                v = bp[(size_t)rr * w + col];
            }
            lds_a[r * LDA + c] = f2bf(v);
        }
        __syncthreads();

        int kmax = A.Kp - k0; if (kmax > 128) kmax = 128;
        for (int kk = 0; kk < kmax; kk += 32) {
            // A fragments for all 4 M-tiles, loaded once per K-step
            ABvec a[4];
#pragma unroll
            for (int mt = 0; mt < 4; ++mt) {
                int base = (mt * 16 + lr) * LDA + kk + hi * 16;
                a[mt].h[0] = *(const v8bf*)&lds_a[base];
                a[mt].h[1] = *(const v8bf*)&lds_a[base + 8];
            }
            int kt = (k0 + kk) >> 5;
            const unsigned short* wp = A.w1p + (size_t)kt * NT1 * 512;
            __builtin_prefetch(wp + (size_t)NT1 * 512 + (size_t)wave * 4 * 512, 0, 0);
#pragma unroll
            for (int t = 0; t < 4; ++t) {
                const v16bf b = *(const v16bf*)(wp +
                    ((size_t)(wave * 4 + t) * 32 + lane) * 16);
#pragma unroll
                for (int mt = 0; mt < 4; ++mt)
                    acc1[mt][t] = __builtin_amdgcn_wmma_f32_16x16x32_bf16(
                        false, a[mt].v, false, b, (short)0, acc1[mt][t], false, false);
            }
        }
    }

    // bias + SiLU -> hidden in LDS (bf16)
#pragma unroll
    for (int t = 0; t < 4; ++t) {
        int ncol = (wave * 4 + t) * 16 + lr;
        float bb = A.b1[ncol];
#pragma unroll
        for (int mt = 0; mt < 4; ++mt)
#pragma unroll
            for (int r = 0; r < 8; ++r) {
                int row = mt * 16 + r + hi * 8;
                float v = acc1[mt][t][r] + bb;
                v = v / (1.0f + __expf(-v));
                lds_h[row * LDH + ncol] = f2bf(v);
            }
    }
    __syncthreads();

    // ---------------- GEMM2: O = H @ W2 + b2 (registers only) --------------
    const int NT2 = A.N2p >> 4;
    int tl[4] = {0, 0, 0, 0};
    int tn = 0;
    for (int t = wave; t < NT2; t += 8) if (tn < 4) tl[tn++] = t;

    v8f acc2[4][4];  // [slot][M-tile]
#pragma unroll
    for (int s = 0; s < 4; ++s)
#pragma unroll
        for (int mt = 0; mt < 4; ++mt) acc2[s][mt] = zero8();

    for (int kk = 0; kk < 512; kk += 32) {
        ABvec a[4];
#pragma unroll
        for (int mt = 0; mt < 4; ++mt) {
            int base = (mt * 16 + lr) * LDH + kk + hi * 16;
            a[mt].h[0] = *(const v8bf*)&lds_h[base];
            a[mt].h[1] = *(const v8bf*)&lds_h[base + 8];
        }
#pragma unroll
        for (int s = 0; s < 4; ++s) {
            if (s < tn) {
                const v16bf b = *(const v16bf*)(A.w2p +
                    ((size_t)((kk >> 5) * NT2 + tl[s]) * 32 + lane) * 16);
#pragma unroll
                for (int mt = 0; mt < 4; ++mt)
                    acc2[s][mt] = __builtin_amdgcn_wmma_f32_16x16x32_bf16(
                        false, a[mt].v, false, b, (short)0, acc2[s][mt], false, false);
            }
        }
    }

    // add bias in-register (before LN stats)
#pragma unroll
    for (int s = 0; s < 4; ++s) {
        if (s < tn) {
            int col = tl[s] * 16 + lr;
            float bb = (col < A.N2real) ? A.b2[col] : 0.0f;
#pragma unroll
            for (int mt = 0; mt < 4; ++mt)
#pragma unroll
                for (int r = 0; r < 8; ++r) acc2[s][mt][r] += bb;
        }
    }

    // ---------------- LayerNorm stats via shuffle + LDS partials -----------
    if (A.gamma) {
#pragma unroll
        for (int mt = 0; mt < 4; ++mt)
#pragma unroll
            for (int r = 0; r < 8; ++r) {
                float s1 = 0.0f, s2 = 0.0f;
#pragma unroll
                for (int s = 0; s < 4; ++s)
                    if (s < tn) { float x = acc2[s][mt][r]; s1 += x; s2 += x * x; }
                // reduce across the 16 lanes of this half-wave (same row)
#pragma unroll
                for (int msk = 1; msk < 16; msk <<= 1) {
                    s1 += __shfl_xor(s1, msk, 32);
                    s2 += __shfl_xor(s2, msk, 32);
                }
                if (lr == 0) {
                    int row = mt * 16 + r + hi * 8;
                    red_s[row * 8 + wave] = s1;
                    red_ss[row * 8 + wave] = s2;
                }
            }
        __syncthreads();
        if (tid < BM) {
            float S = 0.0f, SS = 0.0f;
            for (int w = 0; w < 8; ++w) { S += red_s[tid * 8 + w]; SS += red_ss[tid * 8 + w]; }
            float mean = S * (1.0f / 512.0f);
            float var  = SS * (1.0f / 512.0f) - mean * mean;
            st_mean[tid] = mean;
            st_rstd[tid] = rsqrtf(var + 1e-5f);
        }
        __syncthreads();
    }

    // ---------------- apply LN/res, store, fused atomic segment-sum --------
#pragma unroll
    for (int s = 0; s < 4; ++s) {
        if (s < tn) {
            int col = tl[s] * 16 + lr;
#pragma unroll
            for (int mt = 0; mt < 4; ++mt)
#pragma unroll
                for (int r = 0; r < 8; ++r) {
                    int row = mt * 16 + r + hi * 8;
                    int mm = rowBase + row;
                    if (mm < A.M) {
                        float x = acc2[s][mt][r];
                        if (A.gamma) x = (x - st_mean[row]) * st_rstd[row] * A.gamma[col] + A.beta[col];
                        if (A.res)   x += A.res[(size_t)mm * 512 + col];
                        if (A.outp && col < A.Nout) A.outp[(size_t)mm * A.outStride + col] = x;
                        if (A.aggp)  atomicAdd(&A.aggp[(size_t)A.aggIdx[mm] * 512 + col], x);
                    }
                }
        }
    }
}

// ---------------------------------------------------------------------------
// Host orchestration
// ---------------------------------------------------------------------------
extern "C" void kernel_launch(void* const* d_in, const int* in_sizes, int n_in,
                              void* d_out, int out_size, void* d_ws, size_t ws_size,
                              hipStream_t stream) {
    (void)n_in; (void)out_size; (void)ws_size;
    const float* grid_feat  = (const float*)d_in[0];
    const float* mesh_feat  = (const float*)d_in[1];
    const float* g2m_efeat  = (const float*)d_in[2];
    const float* mesh_efeat = (const float*)d_in[3];
    const float* m2g_efeat  = (const float*)d_in[4];
    const int* g2m_src  = (const int*)d_in[5];
    const int* g2m_dst  = (const int*)d_in[6];
    const int* mesh_src = (const int*)d_in[7];
    const int* mesh_dst = (const int*)d_in[8];
    const int* m2g_src  = (const int*)d_in[9];
    const int* m2g_dst  = (const int*)d_in[10];

    const int n_grid = in_sizes[0] / 474;
    const int n_mesh = in_sizes[1] / 3;
    const int n_g2m  = in_sizes[5];
    const int n_me   = in_sizes[7];
    const int n_m2g  = in_sizes[9];

    char* ws = (char*)d_ws; size_t off = 0;
    auto alloc = [&](size_t b) -> void* {
        void* p = ws + off; off = (off + b + 255) & ~(size_t)255; return p;
    };

    float* gN   = (float*)alloc((size_t)n_grid * 512 * 4);
    float* mN   = (float*)alloc((size_t)n_mesh * 512 * 4);
    float* eM   = (float*)alloc((size_t)n_me   * 512 * 4);
    float* eBig = (float*)alloc((size_t)n_g2m  * 512 * 4);   // reused g2m then m2g
    float* aggM = (float*)alloc((size_t)n_mesh * 512 * 4);
    float* aggG = (float*)alloc((size_t)n_grid * 512 * 4);

    auto PF = [&](int i) { return (const float*)d_in[i]; };

    auto packW = [&](const float* W, int K, int N, int Kp, int Np) -> unsigned short* {
        size_t elems = (size_t)Kp * Np;
        unsigned short* o = (unsigned short*)alloc(elems * 2);
        int nb = (int)((elems + 2047) / 2048); if (nb > 2048) nb = 2048;
        hipLaunchKernelGGL(pack_w, dim3(nb), dim3(256), 0, stream, W, o, K, N, Kp, Np);
        return o;
    };
    auto fillz = [&](float* p, size_t n) {
        int nb = (int)((n + 2047) / 2048); if (nb > 2048) nb = 2048;
        hipLaunchKernelGGL(fill0, dim3(nb), dim3(256), 0, stream, p, n);
    };

    struct PM { unsigned short *w1, *w2; const float *b1, *b2, *ga, *be; int Kp, N2p, N2real; };
    auto mkMlp = [&](int base, int K1) -> PM {
        PM p;
        int Kp = (K1 + 31) & ~31;
        p.w1 = packW(PF(base + 0), K1, 512, Kp, 512);
        p.w2 = packW(PF(base + 1), 512, 512, 512, 512);
        p.b1 = PF(base + 2); p.b2 = PF(base + 3);
        p.be = PF(base + 4); p.ga = PF(base + 5);
        p.Kp = Kp; p.N2p = 512; p.N2real = 512;
        return p;
    };

    auto run = [&](const PM& p, int M,
                   const float* s0, const int* i0, int w0,
                   const float* s1, const int* i1, int w1,
                   const float* s2, const int* i2, int w2,
                   const float* res, float* outp, int Nout, int ostride,
                   float* aggp, const int* aggIdx) {
        MlpArgs a;
        a.s0 = s0; a.i0 = i0; a.w0 = w0;
        a.s1 = s1; a.i1 = i1; a.w1 = w1;
        a.s2 = s2; a.i2 = i2; a.w2 = w2;
        a.M = M; a.Ktot = w0 + w1 + w2; a.Kp = p.Kp;
        a.w1p = p.w1; a.b1 = p.b1; a.w2p = p.w2; a.b2 = p.b2;
        a.gamma = p.ga; a.beta = p.be; a.res = res;
        a.outp = outp; a.Nout = Nout; a.outStride = ostride;
        a.aggp = aggp; a.aggIdx = aggIdx;
        a.N2p = p.N2p; a.N2real = p.N2real;
        hipLaunchKernelGGL(mlp_fused, dim3((M + BM - 1) / BM), dim3(256), 0, stream, a);
    };

    // ---- pack all weights (param leaves flattened in sorted-dict order) ----
    PM dec_edge   = mkMlp(11, 1536);
    PM dec_node   = mkMlp(17, 1024);
    PM emb_g2m    = mkMlp(23, 4);
    PM emb_grid   = mkMlp(29, 474);
    PM emb_m2g    = mkMlp(35, 4);
    PM emb_mesh   = mkMlp(41, 3);
    PM emb_mesh_e = mkMlp(47, 4);
    PM enc_dst    = mkMlp(53, 1024);
    PM enc_edge   = mkMlp(59, 1536);
    PM enc_src    = mkMlp(65, 512);
    PM finale;
    finale.w1 = packW(PF(71), 512, 512, 512, 512);
    finale.w2 = packW(PF(72), 512, 227, 512, 240);
    finale.b1 = PF(73); finale.b2 = PF(74);
    finale.ga = nullptr; finale.be = nullptr;
    finale.Kp = 512; finale.N2p = 240; finale.N2real = 227;

    const float* peW1 = PF(75); const float* peW2 = PF(76);
    const float* peb1 = PF(77); const float* peb2 = PF(78);
    const float* pebe = PF(79); const float* peg  = PF(80);
    const float* pnW1 = PF(81); const float* pnW2 = PF(82);
    const float* pnb1 = PF(83); const float* pnb2 = PF(84);
    const float* pnbe = PF(85); const float* png  = PF(86);

    PM pedge[16], pnode[16];
    for (int l = 0; l < 16; ++l) {
        pedge[l].w1 = packW(peW1 + (size_t)l * 1536 * 512, 1536, 512, 1536, 512);
        pedge[l].w2 = packW(peW2 + (size_t)l * 512 * 512, 512, 512, 512, 512);
        pedge[l].b1 = peb1 + (size_t)l * 512; pedge[l].b2 = peb2 + (size_t)l * 512;
        pedge[l].be = pebe + (size_t)l * 512; pedge[l].ga = peg  + (size_t)l * 512;
        pedge[l].Kp = 1536; pedge[l].N2p = 512; pedge[l].N2real = 512;
        pnode[l].w1 = packW(pnW1 + (size_t)l * 1024 * 512, 1024, 512, 1024, 512);
        pnode[l].w2 = packW(pnW2 + (size_t)l * 512 * 512, 512, 512, 512, 512);
        pnode[l].b1 = pnb1 + (size_t)l * 512; pnode[l].b2 = pnb2 + (size_t)l * 512;
        pnode[l].be = pnbe + (size_t)l * 512; pnode[l].ga = png  + (size_t)l * 512;
        pnode[l].Kp = 1024; pnode[l].N2p = 512; pnode[l].N2real = 512;
    }

    // ---------------------------- pipeline ---------------------------------
    run(emb_grid,   n_grid, grid_feat,  nullptr, 474, nullptr, nullptr, 0, nullptr, nullptr, 0,
        nullptr, gN,   512, 512, nullptr, nullptr);
    run(emb_mesh,   n_mesh, mesh_feat,  nullptr, 3,   nullptr, nullptr, 0, nullptr, nullptr, 0,
        nullptr, mN,   512, 512, nullptr, nullptr);
    run(emb_g2m,    n_g2m,  g2m_efeat,  nullptr, 4,   nullptr, nullptr, 0, nullptr, nullptr, 0,
        nullptr, eBig, 512, 512, nullptr, nullptr);
    run(emb_mesh_e, n_me,   mesh_efeat, nullptr, 4,   nullptr, nullptr, 0, nullptr, nullptr, 0,
        nullptr, eM,   512, 512, nullptr, nullptr);

    // g2m encoder: edge block (+res, fused segment-sum into aggM), node blocks
    fillz(aggM, (size_t)n_mesh * 512);
    run(enc_edge, n_g2m, eBig, nullptr, 512, gN, g2m_src, 512, mN, g2m_dst, 512,
        eBig, nullptr, 0, 0, aggM, g2m_dst);
    run(enc_dst, n_mesh, aggM, nullptr, 512, mN, nullptr, 512, nullptr, nullptr, 0,
        mN, mN, 512, 512, nullptr, nullptr);
    run(enc_src, n_grid, gN, nullptr, 512, nullptr, nullptr, 0, nullptr, nullptr, 0,
        gN, gN, 512, 512, nullptr, nullptr);

    // 16-layer mesh processor
    for (int l = 0; l < 16; ++l) {
        fillz(aggM, (size_t)n_mesh * 512);
        run(pedge[l], n_me, eM, nullptr, 512, mN, mesh_src, 512, mN, mesh_dst, 512,
            eM, eM, 512, 512, aggM, mesh_dst);
        run(pnode[l], n_mesh, aggM, nullptr, 512, mN, nullptr, 512, nullptr, nullptr, 0,
            mN, mN, 512, 512, nullptr, nullptr);
    }

    // m2g decoder (eBig reused for e_m2g)
    run(emb_m2g, n_m2g, m2g_efeat, nullptr, 4, nullptr, nullptr, 0, nullptr, nullptr, 0,
        nullptr, eBig, 512, 512, nullptr, nullptr);
    fillz(aggG, (size_t)n_grid * 512);
    run(dec_edge, n_m2g, eBig, nullptr, 512, mN, m2g_src, 512, gN, m2g_dst, 512,
        eBig, nullptr, 0, 0, aggG, m2g_dst);
    run(dec_node, n_grid, aggG, nullptr, 512, gN, nullptr, 512, nullptr, nullptr, 0,
        gN, gN, 512, 512, nullptr, nullptr);

    // finale (no norm, no residual) -> d_out [n_grid, 227]
    run(finale, n_grid, gN, nullptr, 512, nullptr, nullptr, 0, nullptr, nullptr, 0,
        nullptr, (float*)d_out, 227, 227, nullptr, nullptr);
}
